// RecursiveCompressor_10866267259050
// MI455X (gfx1250) — matrix operand order repaired and checked by reading, not compiled, not run on hardware
//
#include <hip/hip_runtime.h>
#include <hip/hip_bf16.h>
#include <cstdint>
#include <cstddef>

// ---------------- model constants (static shapes) ----------------
#define DMODEL  1024
#define NHEAD   16
#define HDIM    64
#define DFFN    2048
#define CHUNKSZ 64
#define CSZ     8
#define BSZ     4
#define SEQ     4096

typedef __attribute__((ext_vector_type(16))) _Float16 v16h;
typedef __attribute__((ext_vector_type(8)))  _Float16 v8h;
typedef __attribute__((ext_vector_type(8)))  float    v8f;
typedef __attribute__((ext_vector_type(4)))  unsigned int v4u;
typedef __attribute__((ext_vector_type(8)))  int      v8i;
typedef __attribute__((ext_vector_type(4)))  int      v4i;

// LDS row pitch for staged tiles: 32 f16 (64B) + 16B TDM pad = 80B = 40 halves
// (20 DWORDs -> 16 consecutive rows hit 16 distinct bank groups).
#define LDS_PITCH 40

__device__ inline v16h cat8h(v8h lo, v8h hi) {
  return __builtin_shufflevector(lo, hi, 0, 1, 2, 3, 4, 5, 6, 7,
                                         8, 9, 10, 11, 12, 13, 14, 15);
}

// Generic->LDS offset: LDS aperture keeps the byte offset in addr[31:0].
__device__ inline unsigned lds_offset(const void* p) {
  return (unsigned)(size_t)p;
}

// =================================================================
// Tensor Data Mover: async 2D tile load (rows x 32 f16) into LDS.
// D# per CDNA5 ISA 8.3/8.4: group0 = {count, lds_addr, global_addr, type=2},
// group1 = {data_size=2B, pad(16B per 64B row), dims, tile dims, stride}.
// Tracked on TENSORcnt; issued by wave 0, published via barrier.
// =================================================================
__device__ inline void tdm_load_tile(const _Float16* gsrc, unsigned ldsoff,
                                     unsigned tile_rows, unsigned tensor_d0,
                                     unsigned tensor_d1, unsigned stride_elems) {
  unsigned long long ga = (unsigned long long)(size_t)gsrc;
  v4u g0;
  g0[0] = 1u;                                   // count=1 (valid user D#)
  g0[1] = ldsoff;                               // lds_addr
  g0[2] = (unsigned)(ga & 0xFFFFFFFFu);         // global_addr[31:0]
  g0[3] = (unsigned)((ga >> 32) & 0x01FFFFFFu)  // global_addr[56:32]
        | (2u << 30);                           // type=2 ("image")
  v8i g1;
  g1[0] = (int)((1u << 16)      // data_size = 2 bytes
              | (1u << 20)      // pad_enable
              | (3u << 22)      // pad_interval: every 16 DWORDs (=64B row)
              | (3u << 25));    // pad_amount: 4 DWORDs (=16B) -> 80B pitch
  g1[1] = (int)((tensor_d0 & 0xFFFFu) << 16);                     // dim0[15:0]
  g1[2] = (int)(((tensor_d0 >> 16) & 0xFFFFu)
              | ((tensor_d1 & 0xFFFFu) << 16));                   // dim0 hi, dim1 lo
  g1[3] = (int)(((tensor_d1 >> 16) & 0xFFFFu) | (32u << 16));     // dim1 hi, tile_dim0=32
  g1[4] = (int)(tile_rows & 0xFFFFu);                             // tile_dim1
  g1[5] = (int)stride_elems;                                      // dim0_stride[31:0]
  g1[6] = 0;
  g1[7] = 0;
  v4i z4 = {0, 0, 0, 0};
#if defined(__clang_major__) && (__clang_major__ >= 23)
  v8i z8 = {0, 0, 0, 0, 0, 0, 0, 0};
  __builtin_amdgcn_tensor_load_to_lds(g0, g1, z4, z4, z8, 0);
#else
  __builtin_amdgcn_tensor_load_to_lds(g0, g1, z4, z4, 0);
#endif
}

// =================================================================
// WMMA GEMM:  out[M][N] = A[M][K](f16) @ WT[N][K]^T(f16) + bias (+res)
// grid = (N/64, M/128), block = 256 threads (8 waves).
// Double-buffered TDM pipeline, K-stage = 64 (2 slices of 32):
// wave 0 DMAs 4 tiles (A-lo/hi, B-lo/hi) for the next stage while all
// 8 waves run 8 WMMAs on the current buffer.
// Compile-time epilogue variants avoid per-element branching.
// =================================================================
template <bool HASRES, bool O32, bool O16>
__global__ __launch_bounds__(256) void gemm_f16_wmma(
    const _Float16* __restrict__ A, const _Float16* __restrict__ WT,
    const float* __restrict__ bias, const float* __restrict__ residual,
    float* __restrict__ out32, _Float16* __restrict__ out16,
    int M, int N, int K) {
  __shared__ _Float16 At[2][2][128 * LDS_PITCH];   // 40 KB
  __shared__ _Float16 Bt[2][2][64 * LDS_PITCH];    // 20 KB

  const int lane  = threadIdx.x & 31;
  const int wave  = threadIdx.x >> 5;
  const int m_blk = blockIdx.y * 128;
  const int n_blk = blockIdx.x * 64;
  const bool issuer = (threadIdx.x < 32);   // wave 0 drives the TDM

  auto stage = [&](int sb, int kk) {
    tdm_load_tile(A  + (size_t)m_blk * K + kk,      lds_offset(&At[sb][0][0]), 128, K, M, K);
    tdm_load_tile(A  + (size_t)m_blk * K + kk + 32, lds_offset(&At[sb][1][0]), 128, K, M, K);
    tdm_load_tile(WT + (size_t)n_blk * K + kk,      lds_offset(&Bt[sb][0][0]),  64, K, N, K);
    tdm_load_tile(WT + (size_t)n_blk * K + kk + 32, lds_offset(&Bt[sb][1][0]),  64, K, N, K);
  };

  if (issuer) {
    stage(0, 0);
    __builtin_amdgcn_s_wait_tensorcnt((short)0);
  }
  __syncthreads();

  v8f acc[4] = {};
  int buf = 0;
  for (int kk = 0; kk < K; kk += 64) {
    if ((kk + 64 < K) && issuer) stage(buf ^ 1, kk + 64);

#pragma unroll
    for (int s = 0; s < 2; ++s) {
      // A 16x32 f16 fragment: lanes 0-15 -> K {0..15},
      // lanes 16-31 -> K {16..31}; row M = lane%16.
      const int mrow = wave * 16 + (lane & 15);
      const int aoff = (lane < 16) ? 0 : 16;
      const _Float16* ap = &At[buf][s][mrow * LDS_PITCH];
      v16h a = cat8h(*(const v8h*)(ap + aoff), *(const v8h*)(ap + aoff + 8));

      // B 32x16 fragments: lane n%16 holds column n, contiguous K-run.
      const int koff = (lane < 16) ? 0 : 16;
      const _Float16* bp0 = &Bt[buf][s][(lane & 15) * LDS_PITCH + koff];
      const _Float16* bp1 = bp0 + 16 * LDS_PITCH;
      const _Float16* bp2 = bp0 + 32 * LDS_PITCH;
      const _Float16* bp3 = bp0 + 48 * LDS_PITCH;
      v16h b0 = cat8h(*(const v8h*)bp0, *(const v8h*)(bp0 + 8));
      v16h b1 = cat8h(*(const v8h*)bp1, *(const v8h*)(bp1 + 8));
      v16h b2 = cat8h(*(const v8h*)bp2, *(const v8h*)(bp2 + 8));
      v16h b3 = cat8h(*(const v8h*)bp3, *(const v8h*)(bp3 + 8));

      acc[0] = __builtin_amdgcn_wmma_f32_16x16x32_f16(false, a, false, b0, (short)0, acc[0], false, false);
      acc[1] = __builtin_amdgcn_wmma_f32_16x16x32_f16(false, a, false, b1, (short)0, acc[1], false, false);
      acc[2] = __builtin_amdgcn_wmma_f32_16x16x32_f16(false, a, false, b2, (short)0, acc[2], false, false);
      acc[3] = __builtin_amdgcn_wmma_f32_16x16x32_f16(false, a, false, b3, (short)0, acc[3], false, false);
    }

    if (issuer) __builtin_amdgcn_s_wait_tensorcnt((short)0);
    __syncthreads();
    buf ^= 1;
  }

  // C/D layout: VGPR j -> M = m_base + j (+8 for lanes 16-31); N = lane%16.
  const int m_base = m_blk + wave * 16;
#pragma unroll
  for (int t = 0; t < 4; ++t) {
    const int n = n_blk + t * 16 + (lane & 15);
    const float bv = bias[n];
#pragma unroll
    for (int j = 0; j < 8; ++j) {
      const int m = m_base + j + ((lane & 16) ? 8 : 0);
      const size_t idx = (size_t)m * N + n;
      float vv = acc[t][j] + bv;
      if (HASRES) vv += residual[idx];
      if (O32) out32[idx] = vv;
      if (O16) out16[idx] = (_Float16)vv;
    }
  }
}

// =================================================================
// LayerNorm over D=1024, f32 in -> f16 out. One block per row.
// =================================================================
__global__ __launch_bounds__(256) void ln_f16_kernel(
    const float* __restrict__ x, const float* __restrict__ g,
    const float* __restrict__ bta, _Float16* __restrict__ out) {
  __shared__ float r1[256];
  __shared__ float r2[256];
  const size_t row = blockIdx.x;
  const float4 vv = ((const float4*)(x + row * DMODEL))[threadIdx.x];
  float s  = vv.x + vv.y + vv.z + vv.w;
  float s2 = vv.x * vv.x + vv.y * vv.y + vv.z * vv.z + vv.w * vv.w;
  r1[threadIdx.x] = s; r2[threadIdx.x] = s2;
  __syncthreads();
  for (int off = 128; off > 0; off >>= 1) {
    if (threadIdx.x < (unsigned)off) {
      r1[threadIdx.x] += r1[threadIdx.x + off];
      r2[threadIdx.x] += r2[threadIdx.x + off];
    }
    __syncthreads();
  }
  const float mean = r1[0] * (1.0f / DMODEL);
  const float var  = r2[0] * (1.0f / DMODEL) - mean * mean;
  const float inv  = rsqrtf(var + 1e-5f);
  const int c = threadIdx.x * 4;
  _Float16* o = out + row * DMODEL + c;
  o[0] = (_Float16)((vv.x - mean) * inv * g[c + 0] + bta[c + 0]);
  o[1] = (_Float16)((vv.y - mean) * inv * g[c + 1] + bta[c + 1]);
  o[2] = (_Float16)((vv.z - mean) * inv * g[c + 2] + bta[c + 2]);
  o[3] = (_Float16)((vv.w - mean) * inv * g[c + 3] + bta[c + 3]);
}

// =================================================================
// Attention core: one block per (chunk, head); blockDim.x = SQ.
// =================================================================
template <int SK, bool CAUSAL>
__global__ __launch_bounds__(64) void attn_kernel(
    const _Float16* __restrict__ q, const _Float16* __restrict__ k,
    const _Float16* __restrict__ v, _Float16* __restrict__ out) {
  __shared__ _Float16 Qs[64 * HDIM];
  __shared__ float    Ks[SK * HDIM];
  __shared__ float    Vs[SK * HDIM];
  __shared__ float    Ss[64 * SK];
  const int chunk = blockIdx.x;
  const int h     = blockIdx.y;
  const int SQ    = blockDim.x;
  const int qi    = threadIdx.x;

  const _Float16* qb = q + ((size_t)chunk * SQ) * DMODEL + h * HDIM;
  const _Float16* kb = k + ((size_t)chunk * SK) * DMODEL + h * HDIM;
  const _Float16* vb = v + ((size_t)chunk * SK) * DMODEL + h * HDIM;

  for (int t = qi; t < SQ * HDIM; t += SQ)
    Qs[t] = qb[(size_t)(t >> 6) * DMODEL + (t & 63)];
  for (int t = qi; t < SK * HDIM; t += SQ) {
    Ks[t] = (float)kb[(size_t)(t >> 6) * DMODEL + (t & 63)];
    Vs[t] = (float)vb[(size_t)(t >> 6) * DMODEL + (t & 63)];
  }
  __syncthreads();

  const _Float16* myq = Qs + qi * HDIM;
  float* mys = Ss + qi * SK;
  const int jmax = CAUSAL ? (qi + 1) : SK;

  float mx = -3.0e38f;
  for (int j = 0; j < jmax; ++j) {
    float s = 0.0f;
    for (int d = 0; d < HDIM; ++d) s += (float)myq[d] * Ks[j * HDIM + d];
    s *= 0.125f;   // 1/sqrt(64)
    mys[j] = s;
    mx = fmaxf(mx, s);
  }
  float sum = 0.0f;
  for (int j = 0; j < jmax; ++j) {
    float e = __expf(mys[j] - mx);
    mys[j] = e;
    sum += e;
  }
  const float rs = 1.0f / sum;
  _Float16* orow = out + ((size_t)chunk * SQ + qi) * DMODEL + h * HDIM;
  for (int d = 0; d < HDIM; ++d) {
    float acc = 0.0f;
    for (int j = 0; j < jmax; ++j) acc += mys[j] * Vs[j * HDIM + d];
    orow[d] = (_Float16)(acc * rs);
  }
}

// =================================================================
// SwiGLU: h[r][4096] -> out16[r][2048] = silu(h[:, :2048]) * h[:, 2048:]
// =================================================================
__global__ __launch_bounds__(256) void swiglu_kernel(
    const float* __restrict__ h, _Float16* __restrict__ out, int rows) {
  const size_t idx = (size_t)blockIdx.x * blockDim.x + threadIdx.x;
  const size_t total = (size_t)rows * DFFN;
  if (idx >= total) return;
  const size_t r = idx / DFFN, c = idx % DFFN;
  const float a  = h[r * (2 * DFFN) + c];
  const float gg = h[r * (2 * DFFN) + DFFN + c];
  out[idx] = (_Float16)((a / (1.0f + __expf(-a))) * gg);
}

// =================================================================
// helpers: transpose+convert weights, converts, broadcasts, permutes
// =================================================================
__global__ __launch_bounds__(256) void transpose_f16_kernel(
    const float* __restrict__ W, _Float16* __restrict__ WT, int K, int N) {
  __shared__ float tile[32][33];
  const int nb = blockIdx.x * 32, kb = blockIdx.y * 32;
  const int tx = threadIdx.x, ty = threadIdx.y;
  for (int i = ty; i < 32; i += 8)
    tile[i][tx] = W[(size_t)(kb + i) * N + nb + tx];
  __syncthreads();
  for (int i = ty; i < 32; i += 8)
    WT[(size_t)(nb + i) * K + kb + tx] = (_Float16)tile[tx][i];
}

__global__ __launch_bounds__(256) void f32_to_f16_kernel(
    const float* __restrict__ in, _Float16* __restrict__ out, size_t n) {
  const size_t idx = (size_t)blockIdx.x * blockDim.x + threadIdx.x;
  if (idx < n) out[idx] = (_Float16)in[idx];
}

// dst row r (r < rows) = src row (r % 8)  [compressor query broadcast]
__global__ __launch_bounds__(256) void bcast_f16_kernel(
    const _Float16* __restrict__ src, _Float16* __restrict__ dst, int rows) {
  const size_t idx = (size_t)blockIdx.x * blockDim.x + threadIdx.x;
  if (idx >= (size_t)rows * DMODEL) return;
  dst[idx] = src[((idx >> 10) % CSZ) * DMODEL + (idx & 1023)];
}
__global__ __launch_bounds__(256) void bcast_f32_kernel(
    const float* __restrict__ src, float* __restrict__ dst, int rows) {
  const size_t idx = (size_t)blockIdx.x * blockDim.x + threadIdx.x;
  if (idx >= (size_t)rows * DMODEL) return;
  dst[idx] = src[((idx >> 10) % CSZ) * DMODEL + (idx & 1023)];
}

// level-0 compressor out [(b*64+chunk)*8 + c] -> level-1 rows [(b*8+c)*64 + chunk]
__global__ __launch_bounds__(256) void permute_comp_kernel(
    const float* __restrict__ src, float* __restrict__ dst) {
  const size_t idx = (size_t)blockIdx.x * blockDim.x + threadIdx.x;
  const size_t total = (size_t)BSZ * 64 * CSZ * DMODEL;
  if (idx >= total) return;
  const int d = idx & 1023;
  const size_t r = idx >> 10;
  const int c = r & 7;
  const size_t t = r >> 3;
  const int chunk = t & 63;
  const int b = (int)(t >> 6);
  dst[((((size_t)(b * CSZ + c)) * 64 + chunk) << 10) + d] = src[idx];
}

// comp = concat([cq, rec[:, :-1]]):
// out[(b*64+chunk)*8+c] = chunk==0 ? cq[c] : xb1[(b*8+c)*64 + chunk-1]
__global__ __launch_bounds__(256) void build_shift_kernel(
    const float* __restrict__ xb1, const float* __restrict__ cq,
    float* __restrict__ out) {
  const size_t idx = (size_t)blockIdx.x * blockDim.x + threadIdx.x;
  const size_t total = (size_t)BSZ * 64 * CSZ * DMODEL;
  if (idx >= total) return;
  const int d = idx & 1023;
  const size_t r = idx >> 10;
  const int c = r & 7;
  const size_t t = r >> 3;
  const int chunk = t & 63;
  const int b = (int)(t >> 6);
  out[idx] = (chunk == 0)
      ? cq[(size_t)c * DMODEL + d]
      : xb1[((((size_t)(b * CSZ + c)) * 64 + (chunk - 1)) << 10) + d];
}

// =================================================================
// Host orchestration
// =================================================================
struct Lin  { const float* W; const float* b; int K; int N; _Float16* WT; };
struct MhaP { Lin q, k, v, o; };
struct FfnP { Lin l1, l2; };
struct NormP { const float* g; const float* b; };

extern "C" void kernel_launch(void* const* d_in, const int* in_sizes, int n_in,
                              void* d_out, int out_size, void* d_ws, size_t ws_size,
                              hipStream_t stream) {
  (void)in_sizes; (void)n_in; (void)out_size; (void)ws_size;

  const int R0  = BSZ * SEQ;                   // 16384 tokens, level 0
  const int CH0 = R0 / CHUNKSZ;                // 256 chunks
  const int R1  = BSZ * CSZ * (SEQ / CHUNKSZ); // 2048 tokens, level 1
  const int CH1 = R1 / CHUNKSZ;                // 32 chunks
  const int FFN_TILE = 4096;

  // ---- inputs (setup_inputs dict order, depth-first) ----
  int ii = 0;
  const float* x  = (const float*)d_in[ii++];
  const float* cq = (const float*)d_in[ii++];
  auto readLin  = [&](int K, int N) { Lin L; L.W = (const float*)d_in[ii++];
                                      L.b = (const float*)d_in[ii++];
                                      L.K = K; L.N = N; L.WT = nullptr; return L; };
  auto readMha  = [&]() { MhaP m; m.q = readLin(DMODEL, DMODEL);
                          m.k = readLin(DMODEL, DMODEL); m.v = readLin(DMODEL, DMODEL);
                          m.o = readLin(DMODEL, DMODEL); return m; };
  MhaP mEnc = readMha(), mDec = readMha(), mCmp = readMha(), mDcmp = readMha();
  auto readFfn  = [&]() { FfnP f; f.l1 = readLin(DMODEL, 2 * DFFN);
                          f.l2 = readLin(DFFN, DMODEL); return f; };
  FfnP fEnc = readFfn(), fDec = readFfn();
  auto readNorm = [&]() { NormP n; n.g = (const float*)d_in[ii++];
                          n.b = (const float*)d_in[ii++]; return n; };
  NormP nMhaE = readNorm(), nFfnE = readNorm(), nMhaD = readNorm(),
        nFfnD = readNorm(), nCmp = readNorm(), nDcmp = readNorm();

  // ---- workspace bump allocator ----
  uint8_t* wsp = (uint8_t*)d_ws;
  auto alloc = [&](size_t bytes) -> void* {
    void* r = (void*)wsp;
    wsp += (bytes + 255) & ~(size_t)255;
    return r;
  };

  // ---- weight prep: f32 [K][N] -> f16 [N][K] ----
  auto prepLin = [&](Lin& L) {
    L.WT = (_Float16*)alloc((size_t)L.K * L.N * sizeof(_Float16));
    transpose_f16_kernel<<<dim3(L.N / 32, L.K / 32), dim3(32, 8), 0, stream>>>(
        L.W, L.WT, L.K, L.N);
  };
  Lin* all[] = {&mEnc.q,&mEnc.k,&mEnc.v,&mEnc.o, &mDec.q,&mDec.k,&mDec.v,&mDec.o,
                &mCmp.q,&mCmp.k,&mCmp.v,&mCmp.o, &mDcmp.q,&mDcmp.k,&mDcmp.v,&mDcmp.o,
                &fEnc.l1,&fEnc.l2, &fDec.l1,&fDec.l2};
  for (Lin* L : all) prepLin(*L);

  _Float16* cq16 = (_Float16*)alloc((size_t)CSZ * DMODEL * sizeof(_Float16));
  f32_to_f16_kernel<<<(CSZ * DMODEL + 255) / 256, 256, 0, stream>>>(
      cq, cq16, (size_t)CSZ * DMODEL);

  // ---- activation buffers ----
  float*    xb0     = (float*)alloc((size_t)R0 * DMODEL * sizeof(float));
  float*    xb1     = (float*)alloc((size_t)R1 * DMODEL * sizeof(float));
  _Float16* tmp16   = (_Float16*)alloc((size_t)R0 * DMODEL * sizeof(_Float16));
  _Float16* xq16    = (_Float16*)alloc((size_t)R0 * DMODEL * sizeof(_Float16));
  _Float16* q16     = (_Float16*)alloc((size_t)R0 * DMODEL * sizeof(_Float16));
  _Float16* k16     = (_Float16*)alloc((size_t)R0 * DMODEL * sizeof(_Float16));
  _Float16* v16     = (_Float16*)alloc((size_t)R0 * DMODEL * sizeof(_Float16));
  _Float16* ctx16   = (_Float16*)alloc((size_t)R0 * DMODEL * sizeof(_Float16));
  float*    ffn_h   = (float*)alloc((size_t)FFN_TILE * 2 * DFFN * sizeof(float));
  _Float16* ffn_g16 = (_Float16*)alloc((size_t)FFN_TILE * DFFN * sizeof(_Float16));
  float*    comp0   = (float*)alloc((size_t)CH0 * CSZ * DMODEL * sizeof(float));
  float*    comp_sh = (float*)alloc((size_t)CH0 * CSZ * DMODEL * sizeof(float));
  _Float16* compn16 = (_Float16*)alloc((size_t)CH0 * CSZ * DMODEL * sizeof(_Float16));
  _Float16* cqb16   = (_Float16*)alloc((size_t)CH0 * CSZ * DMODEL * sizeof(_Float16));
  float*    comp1f  = (float*)alloc((size_t)CH1 * CSZ * DMODEL * sizeof(float));
  _Float16* l1q16   = (_Float16*)alloc((size_t)R1 * DMODEL * sizeof(_Float16));

  // ---- launch helpers (compile-time epilogue selection) ----
  auto gemm = [&](const _Float16* A, const Lin& L, const float* residual,
                  float* out32, _Float16* out16, int M, int N, int K) {
    dim3 grd(N / 64, M / 128);
    if (residual)
      gemm_f16_wmma<true, true, false><<<grd, 256, 0, stream>>>(
          A, L.WT, L.b, residual, out32, out16, M, N, K);
    else if (out32)
      gemm_f16_wmma<false, true, false><<<grd, 256, 0, stream>>>(
          A, L.WT, L.b, residual, out32, out16, M, N, K);
    else
      gemm_f16_wmma<false, false, true><<<grd, 256, 0, stream>>>(
          A, L.WT, L.b, residual, out32, out16, M, N, K);
  };

  // pre-norm MHA (causal, self) + SwiGLU FFN block, in-place on xb
  auto run_block = [&](const MhaP& mha, const FfnP& ffn, const NormP& nm,
                       const NormP& nf, float* xb, int rows) {
    const int chunks = rows / CHUNKSZ;
    ln_f16_kernel<<<rows, 256, 0, stream>>>(xb, nm.g, nm.b, tmp16);
    gemm(tmp16, mha.q, nullptr, nullptr, q16, rows, DMODEL, DMODEL);
    gemm(tmp16, mha.k, nullptr, nullptr, k16, rows, DMODEL, DMODEL);
    gemm(tmp16, mha.v, nullptr, nullptr, v16, rows, DMODEL, DMODEL);
    attn_kernel<CHUNKSZ, true><<<dim3(chunks, NHEAD), CHUNKSZ, 0, stream>>>(
        q16, k16, v16, ctx16);
    gemm(ctx16, mha.o, xb, xb, nullptr, rows, DMODEL, DMODEL);
    ln_f16_kernel<<<rows, 256, 0, stream>>>(xb, nf.g, nf.b, tmp16);
    for (int r0 = 0; r0 < rows; r0 += FFN_TILE) {
      const int mr = (rows - r0 < FFN_TILE) ? (rows - r0) : FFN_TILE;
      gemm(tmp16 + (size_t)r0 * DMODEL, ffn.l1, nullptr, ffn_h, nullptr,
           mr, 2 * DFFN, DMODEL);
      const size_t tot = (size_t)mr * DFFN;
      swiglu_kernel<<<(tot + 255) / 256, 256, 0, stream>>>(ffn_h, ffn_g16, mr);
      gemm(ffn_g16, ffn.l2, xb + (size_t)r0 * DMODEL, xb + (size_t)r0 * DMODEL,
           nullptr, mr, DMODEL, DFFN);
    }
  };

  // =========================== level 0 ===========================
  hipMemcpyAsync(xb0, x, (size_t)R0 * DMODEL * sizeof(float),
                 hipMemcpyDeviceToDevice, stream);
  run_block(mEnc, fEnc, nMhaE, nFfnE, xb0, R0);

  // xq = LN(norm_compressor, xb)  (also decompressor q_in later)
  ln_f16_kernel<<<R0, 256, 0, stream>>>(xb0, nCmp.g, nCmp.b, xq16);

  // compressor cross-attention: q = cq broadcast (CH0*CS rows), kv = xq
  {
    const size_t nq = (size_t)CH0 * CSZ * DMODEL;
    bcast_f16_kernel<<<(nq + 255) / 256, 256, 0, stream>>>(cq16, cqb16, CH0 * CSZ);
    gemm(cqb16, mCmp.q, nullptr, nullptr, q16, CH0 * CSZ, DMODEL, DMODEL);
    gemm(xq16, mCmp.k, nullptr, nullptr, k16, R0, DMODEL, DMODEL);
    gemm(xq16, mCmp.v, nullptr, nullptr, v16, R0, DMODEL, DMODEL);
    attn_kernel<CHUNKSZ, false><<<dim3(CH0, NHEAD), CSZ, 0, stream>>>(
        q16, k16, v16, ctx16);
    gemm(ctx16, mCmp.o, nullptr, comp0, nullptr, CH0 * CSZ, DMODEL, DMODEL);
  }

  // reshape/transpose to level-1 sequence layout
  {
    const size_t tot = (size_t)R1 * DMODEL;
    permute_comp_kernel<<<(tot + 255) / 256, 256, 0, stream>>>(comp0, xb1);
  }

  // =========================== level 1 (base case) ===============
  run_block(mEnc, fEnc, nMhaE, nFfnE, xb1, R1);
  {
    const size_t tot = (size_t)R1 * DMODEL;
    f32_to_f16_kernel<<<(tot + 255) / 256, 256, 0, stream>>>(xb1, l1q16, tot);
    const size_t nc = (size_t)CH1 * CSZ * DMODEL;
    bcast_f32_kernel<<<(nc + 255) / 256, 256, 0, stream>>>(cq, comp1f, CH1 * CSZ);
    ln_f16_kernel<<<CH1 * CSZ, 256, 0, stream>>>(comp1f, nDcmp.g, nDcmp.b, compn16);
    gemm(l1q16, mDcmp.q, nullptr, nullptr, q16, R1, DMODEL, DMODEL);
    gemm(compn16, mDcmp.k, nullptr, nullptr, k16, CH1 * CSZ, DMODEL, DMODEL);
    gemm(compn16, mDcmp.v, nullptr, nullptr, v16, CH1 * CSZ, DMODEL, DMODEL);
    attn_kernel<CSZ, false><<<dim3(CH1, NHEAD), CHUNKSZ, 0, stream>>>(
        q16, k16, v16, ctx16);
    gemm(ctx16, mDcmp.o, xb1, xb1, nullptr, R1, DMODEL, DMODEL);
  }
  run_block(mDec, fDec, nMhaD, nFfnD, xb1, R1);

  // =========================== back at level 0 ===================
  {
    const size_t nc = (size_t)CH0 * CSZ * DMODEL;
    build_shift_kernel<<<(nc + 255) / 256, 256, 0, stream>>>(xb1, cq, comp_sh);
    ln_f16_kernel<<<CH0 * CSZ, 256, 0, stream>>>(comp_sh, nDcmp.g, nDcmp.b, compn16);
    gemm(xq16, mDcmp.q, nullptr, nullptr, q16, R0, DMODEL, DMODEL);
    gemm(compn16, mDcmp.k, nullptr, nullptr, k16, CH0 * CSZ, DMODEL, DMODEL);
    gemm(compn16, mDcmp.v, nullptr, nullptr, v16, CH0 * CSZ, DMODEL, DMODEL);
    attn_kernel<CSZ, false><<<dim3(CH0, NHEAD), CHUNKSZ, 0, stream>>>(
        q16, k16, v16, ctx16);
    gemm(ctx16, mDcmp.o, xb0, xb0, nullptr, R0, DMODEL, DMODEL);
  }
  run_block(mDec, fDec, nMhaD, nFfnD, xb0, R0);

  hipMemcpyAsync(d_out, xb0, (size_t)R0 * DMODEL * sizeof(float),
                 hipMemcpyDeviceToDevice, stream);
}
